// CrossSwinBlock_712964571286
// MI455X (gfx1250) — matrix operand order repaired and checked by reading, not compiled, not run on hardware
//
#include <hip/hip_runtime.h>
#include <cstdint>
#include <cstddef>

// ---------------------------------------------------------------------------
// CrossSwinBlock on gfx1250: bf16 WMMA GEMMs (f32 accum), wave32,
// async-to-LDS staging (ASYNCcnt) in the attention kernel.
// ---------------------------------------------------------------------------

typedef __bf16 bf16;
typedef __attribute__((ext_vector_type(16))) __bf16 bf16x16;
typedef __attribute__((ext_vector_type(8)))  __bf16 bf16x8;
typedef __attribute__((ext_vector_type(8)))  float  f32x8;

#define DEV __device__ __forceinline__

// ---- static problem config -------------------------------------------------
static constexpr int Cdim  = 256;     // channels
static constexpr int NHd   = 8;       // heads
static constexpr int HDim  = 32;      // head dim
static constexpr int WSz   = 7;       // window size
static constexpr int Ntok  = 49;      // tokens / window
static constexpr int Npad  = 64;      // padded tokens (WMMA)
static constexpr int Tt    = 4, Hh = 56, Ww = 56;
static constexpr int ImgH  = Tt * Hh; // 224
static constexpr int Mrows = 4 * 256 * Ntok;  // 50176 == B*L
static constexpr int HID   = 1024;

// ---- WMMA fragment helpers (contiguous k order: k = 16*(lane/16) + s, same for A and B) ------------------
DEV bf16x16 load_a_frag(const bf16* A, int lda, int lane) {
  const int m  = lane & 15;
  const int kb = (lane >> 4) << 4;              // 0 or 16
  return *reinterpret_cast<const bf16x16*>(A + (size_t)m * lda + kb);
}
DEV bf16x16 load_b_frag(const bf16* Bt, int ldb, int lane) {  // Bt is N x K row-major
  const int n  = lane & 15;
  const int kh = (lane >> 4) << 4;              // 0 or 16
  return *reinterpret_cast<const bf16x16*>(Bt + (size_t)n * ldb + kh);
}
DEV f32x8 wmma_bf(bf16x16 a, bf16x16 b, f32x8 c) {
  return __builtin_amdgcn_wmma_f32_16x16x32_bf16(false, a, false, b, (short)0, c,
                                                 false, false);
}
// 32-row x 64-col tile per wave: 2 A-frags x 4 B-frags -> 8 WMMAs per k-step
DEV void gemm_core_2x4(const bf16* A, const bf16* Bt, int K, int lane,
                       f32x8 acc[2][4]) {
  for (int ks = 0; ks < K; ks += 32) {
    bf16x16 a0 = load_a_frag(A + ks, K, lane);
    bf16x16 a1 = load_a_frag(A + (size_t)16 * K + ks, K, lane);
#pragma unroll
    for (int j = 0; j < 4; ++j) {
      bf16x16 b = load_b_frag(Bt + (size_t)(j * 16) * K + ks, K, lane);
      acc[0][j] = wmma_bf(a0, b, acc[0][j]);
      acc[1][j] = wmma_bf(a1, b, acc[1][j]);
    }
  }
}

// ---- async global->LDS staging (gfx1250, ASYNCcnt-tracked) -----------------
DEV void async_copy_b128(const void* gsrc, void* lds_dst) {
  const unsigned lds_off = (unsigned)(uintptr_t)lds_dst;       // low 32b = LDS offset
  const unsigned long long ga = (unsigned long long)(uintptr_t)gsrc;
  asm volatile("global_load_async_to_lds_b128 %0, %1, off"
               :: "v"(lds_off), "v"(ga) : "memory");
}
DEV void async_wait0() {
  asm volatile("s_wait_asynccnt 0" ::: "memory");
}

// ---- small utility kernels -------------------------------------------------
__global__ void cvt_f32_bf16_kernel(const float* __restrict__ s, bf16* __restrict__ d, int n) {
  int i = blockIdx.x * 256 + threadIdx.x;
  if (i < n) d[i] = (bf16)s[i];
}
__global__ void zero16_kernel(float4* __restrict__ p, long n) {
  long i = (long)blockIdx.x * blockDim.x + threadIdx.x;
  if (i < n) p[i] = make_float4(0.f, 0.f, 0.f, 0.f);
}

// ---- LN1 + cyclic shift + window partition -> bf16 -------------------------
__global__ void ln1_shift_window_kernel(const float* __restrict__ x,
                                        const float* __restrict__ g,
                                        const float* __restrict__ be,
                                        bf16* __restrict__ xw) {
  const int lane = threadIdx.x & 31;
  const int row  = blockIdx.x * 8 + (threadIdx.x >> 5);   // window-order row
  const int w = row / Ntok, t = row % Ntok;
  const int b = w >> 8, wi = w & 255;
  int hh = (wi >> 3) * WSz + t / WSz;
  int ww = (wi & 7) * WSz + t % WSz;
  int sh = hh + 3; if (sh >= ImgH) sh -= ImgH;            // undo roll(-3)
  int sw = ww + 3; if (sw >= Ww)   sw -= Ww;
  const float* px = x + ((size_t)b * 12544 + sh * Ww + sw) * Cdim + lane * 8;
  float v[8]; float s = 0.f, ss = 0.f;
#pragma unroll
  for (int i = 0; i < 8; ++i) { v[i] = px[i]; s += v[i]; ss += v[i] * v[i]; }
#pragma unroll
  for (int d = 16; d >= 1; d >>= 1) { s += __shfl_xor(s, d, 32); ss += __shfl_xor(ss, d, 32); }
  const float mean = s * (1.f / 256.f);
  const float rstd = rsqrtf(ss * (1.f / 256.f) - mean * mean + 1e-5f);
  const int c0 = lane * 8;
  union { bf16x8 v8; bf16 e[8]; } o;
#pragma unroll
  for (int i = 0; i < 8; ++i)
    o.e[i] = (bf16)((v[i] - mean) * rstd * g[c0 + i] + be[c0 + i]);
  *reinterpret_cast<bf16x8*>(xw + (size_t)row * Cdim + c0) = o.v8;
}

// ---- QKV GEMM + scatter into padded per-(window,head) q/k/vT ---------------
__global__ void qkv_gemm_kernel(const bf16* __restrict__ X, const bf16* __restrict__ Wt,
                                const float* __restrict__ bias,
                                bf16* __restrict__ q, bf16* __restrict__ k,
                                bf16* __restrict__ vT) {
  const int lane = threadIdx.x & 31;
  const int tg   = blockIdx.x * 8 + (threadIdx.x >> 5);
  const int mt = tg / 12, ng = tg % 12;                    // N=768 -> 12 groups of 64
  f32x8 acc[2][4] = {};
  gemm_core_2x4(X + (size_t)mt * 32 * Cdim, Wt + (size_t)ng * 64 * Cdim, Cdim, lane, acc);
  const int nloc = lane & 15, mh = lane >> 4;
#pragma unroll
  for (int mi = 0; mi < 2; ++mi)
#pragma unroll
    for (int j = 0; j < 4; ++j)
#pragma unroll
      for (int r = 0; r < 8; ++r) {
        const int m = mt * 32 + mi * 16 + mh * 8 + r;
        const int c = ng * 64 + j * 16 + nloc;
        const float val = acc[mi][j][r] + bias[c];
        const int w = m / Ntok, t = m % Ntok;
        const int part = c >> 8, head = (c >> 5) & 7, d = c & 31;
        const size_t base = ((size_t)w * NHd + head);
        if (part == 0)
          q[(base * Npad + t) * HDim + d] = (bf16)(val * 0.17677669529663687f);
        else if (part == 1)
          k[(base * Npad + t) * HDim + d] = (bf16)val;
        else
          vT[(base * HDim + d) * Npad + t] = (bf16)val;
      }
}

// ---- windowed attention: S=QK^T + bias + mask, softmax, Y=PV ---------------
// K / V^T tiles staged into LDS with gfx1250 async-to-LDS loads.
__global__ void attn_kernel(const bf16* __restrict__ q, const bf16* __restrict__ k,
                            const bf16* __restrict__ vT, const float* __restrict__ rpb,
                            bf16* __restrict__ yw) {
  __shared__ __align__(32) bf16 sK[Npad * HDim];           // 4 KB
  __shared__ __align__(32) bf16 sV[HDim * Npad];           // 4 KB
  __shared__ __align__(32) bf16 pL[Npad * Npad];           // 8 KB
  const int lane = threadIdx.x & 31, wave = threadIdx.x >> 5;
  const int wh = blockIdx.x;                               // w*8 + head
  const int w = wh >> 3, head = wh & 7;
  const int wi = w & 255;
  const int wr7 = (wi >> 3) * WSz, wc7 = (wi & 7) * WSz;
  const bf16* qh = q  + (size_t)wh * Npad * HDim;
  const bf16* kh = k  + (size_t)wh * Npad * HDim;
  const bf16* vh = vT + (size_t)wh * HDim * Npad;

  // --- async stage K (4KB) and V^T (4KB): 256 x 16B chunks each, 128 thr ----
  const int tid = threadIdx.x;
#pragma unroll
  for (int i = 0; i < 2; ++i) {
    const int c = tid + i * 128;                           // chunk 0..255
    async_copy_b128((const char*)kh + c * 16, (char*)sK + c * 16);
    async_copy_b128((const char*)vh + c * 16, (char*)sV + c * 16);
  }
  async_wait0();
  __syncthreads();

  // S strip: 16 rows x 64 cols, hd=32 is one K-step
  bf16x16 a = load_a_frag(qh + (size_t)wave * 16 * HDim, HDim, lane);
  f32x8 acc[4];
#pragma unroll
  for (int j = 0; j < 4; ++j) {
    bf16x16 b = load_b_frag(sK + (size_t)j * 16 * HDim, HDim, lane);
    f32x8 z = {};
    acc[j] = wmma_bf(a, b, z);
  }
  const int nloc = lane & 15, mh = lane >> 4;
#pragma unroll
  for (int r = 0; r < 8; ++r) {
    const int m = wave * 16 + mh * 8 + r;
    const int mrow = m / WSz, mcol = m % WSz;
    const int hhm = wr7 + mrow, wwm = wc7 + mcol;
    const int cm = (hhm < 217 ? 0 : (hhm < 221 ? 1 : 2)) * 3 +
                   (wwm < 49  ? 0 : (wwm < 53  ? 1 : 2));
    float s[4];
#pragma unroll
    for (int j = 0; j < 4; ++j) {
      const int n = j * 16 + nloc;
      if (m < Ntok && n < Ntok) {
        const int nrow = n / WSz, ncol = n % WSz;
        const int hhn = wr7 + nrow, wwn = wc7 + ncol;
        const int cn = (hhn < 217 ? 0 : (hhn < 221 ? 1 : 2)) * 3 +
                       (wwn < 49  ? 0 : (wwn < 53  ? 1 : 2));
        const int ridx = (mrow - nrow + 6) * 13 + (mcol - ncol + 6);
        s[j] = acc[j][r] + rpb[(size_t)ridx * NHd + head] + (cm != cn ? -100.f : 0.f);
      } else s[j] = -1e30f;
    }
    float mx = fmaxf(fmaxf(s[0], s[1]), fmaxf(s[2], s[3]));
#pragma unroll
    for (int d = 8; d >= 1; d >>= 1) mx = fmaxf(mx, __shfl_xor(mx, d, 32));
    float e[4], sum = 0.f;
#pragma unroll
    for (int j = 0; j < 4; ++j) { e[j] = __expf(s[j] - mx); sum += e[j]; }
#pragma unroll
    for (int d = 8; d >= 1; d >>= 1) sum += __shfl_xor(sum, d, 32);
    const float inv = 1.f / sum;
#pragma unroll
    for (int j = 0; j < 4; ++j)
      pL[m * Npad + j * 16 + nloc] = (bf16)(e[j] * inv);
  }
  __syncthreads();

  // Y = P(64x64) @ V(64x32)
  f32x8 acc2[2] = {};
#pragma unroll
  for (int ks = 0; ks < Npad; ks += 32) {
    bf16x16 a2 = load_a_frag(pL + wave * 16 * Npad + ks, Npad, lane);
#pragma unroll
    for (int j2 = 0; j2 < 2; ++j2) {
      bf16x16 b2 = load_b_frag(sV + (size_t)j2 * 16 * Npad + ks, Npad, lane);
      acc2[j2] = wmma_bf(a2, b2, acc2[j2]);
    }
  }
#pragma unroll
  for (int j2 = 0; j2 < 2; ++j2)
#pragma unroll
    for (int r = 0; r < 8; ++r) {
      const int m = wave * 16 + mh * 8 + r;
      if (m < Ntok) {
        const int c = head * HDim + j2 * 16 + nloc;
        yw[((size_t)w * Ntok + m) * Cdim + c] = (bf16)acc2[j2][r];
      }
    }
}

// ---- proj GEMM + window-reverse + un-shift + residual; also writes `out` ---
__global__ void proj_gemm_kernel(const bf16* __restrict__ Y, const bf16* __restrict__ Wt,
                                 const float* __restrict__ bias,
                                 const float* __restrict__ x1,
                                 float* __restrict__ x1a, float* __restrict__ out2) {
  const int lane = threadIdx.x & 31;
  const int tg = blockIdx.x * 8 + (threadIdx.x >> 5);
  const int mt = tg / 4, ng = tg % 4;                       // N=256 -> 4 groups
  f32x8 acc[2][4] = {};
  gemm_core_2x4(Y + (size_t)mt * 32 * Cdim, Wt + (size_t)ng * 64 * Cdim, Cdim, lane, acc);
  const int nloc = lane & 15, mh = lane >> 4;
#pragma unroll
  for (int mi = 0; mi < 2; ++mi)
#pragma unroll
    for (int j = 0; j < 4; ++j)
#pragma unroll
      for (int r = 0; r < 8; ++r) {
        const int m = mt * 32 + mi * 16 + mh * 8 + r;
        const int c = ng * 64 + j * 16 + nloc;
        const float val = acc[mi][j][r] + bias[c];
        const int w = m / Ntok, t = m % Ntok;
        const int b = w >> 8, wi = w & 255;
        int hh = (wi >> 3) * WSz + t / WSz;
        int ww = (wi & 7) * WSz + t % WSz;
        int sh = hh + 3; if (sh >= ImgH) sh -= ImgH;        // roll(+3) back
        int sw = ww + 3; if (sw >= Ww)   sw -= Ww;
        const size_t idx = ((size_t)b * 12544 + sh * Ww + sw) * Cdim + c;
        out2[idx] = val;
        x1a[idx] = x1[idx] + val;
      }
}

// ---- LN2 (identity mapping) -> bf16 ----------------------------------------
__global__ void ln2_kernel(const float* __restrict__ x, const float* __restrict__ g,
                           const float* __restrict__ be, bf16* __restrict__ h) {
  const int lane = threadIdx.x & 31;
  const int row  = blockIdx.x * 8 + (threadIdx.x >> 5);
  const float* px = x + (size_t)row * Cdim + lane * 8;
  float v[8]; float s = 0.f, ss = 0.f;
#pragma unroll
  for (int i = 0; i < 8; ++i) { v[i] = px[i]; s += v[i]; ss += v[i] * v[i]; }
#pragma unroll
  for (int d = 16; d >= 1; d >>= 1) { s += __shfl_xor(s, d, 32); ss += __shfl_xor(ss, d, 32); }
  const float mean = s * (1.f / 256.f);
  const float rstd = rsqrtf(ss * (1.f / 256.f) - mean * mean + 1e-5f);
  const int c0 = lane * 8;
  union { bf16x8 v8; bf16 e[8]; } o;
#pragma unroll
  for (int i = 0; i < 8; ++i)
    o.e[i] = (bf16)((v[i] - mean) * rstd * g[c0 + i] + be[c0 + i]);
  *reinterpret_cast<bf16x8*>(h + (size_t)row * Cdim + c0) = o.v8;
}

// ---- fc1 GEMM + exact GELU -> bf16 -----------------------------------------
__global__ void fc1_gemm_kernel(const bf16* __restrict__ X, const bf16* __restrict__ Wt,
                                const float* __restrict__ bias, bf16* __restrict__ hid) {
  const int lane = threadIdx.x & 31;
  const int tg = blockIdx.x * 8 + (threadIdx.x >> 5);
  const int mt = tg / 16, ng = tg % 16;                     // N=1024 -> 16 groups
  f32x8 acc[2][4] = {};
  gemm_core_2x4(X + (size_t)mt * 32 * Cdim, Wt + (size_t)ng * 64 * Cdim, Cdim, lane, acc);
  const int nloc = lane & 15, mh = lane >> 4;
#pragma unroll
  for (int mi = 0; mi < 2; ++mi)
#pragma unroll
    for (int j = 0; j < 4; ++j)
#pragma unroll
      for (int r = 0; r < 8; ++r) {
        const int m = mt * 32 + mi * 16 + mh * 8 + r;
        const int c = ng * 64 + j * 16 + nloc;
        const float v = acc[mi][j][r] + bias[c];
        const float ge = 0.5f * v * (1.f + erff(v * 0.70710678118654752f));
        hid[(size_t)m * HID + c] = (bf16)ge;
      }
}

// ---- fc2 GEMM + residual -> x1_new (d_out first half) ----------------------
__global__ void fc2_gemm_kernel(const bf16* __restrict__ X, const bf16* __restrict__ Wt,
                                const float* __restrict__ bias,
                                const float* __restrict__ x1a, float* __restrict__ out1) {
  const int lane = threadIdx.x & 31;
  const int tg = blockIdx.x * 8 + (threadIdx.x >> 5);
  const int mt = tg / 4, ng = tg % 4;                       // N=256, K=1024
  f32x8 acc[2][4] = {};
  gemm_core_2x4(X + (size_t)mt * 32 * HID, Wt + (size_t)ng * 64 * HID, HID, lane, acc);
  const int nloc = lane & 15, mh = lane >> 4;
#pragma unroll
  for (int mi = 0; mi < 2; ++mi)
#pragma unroll
    for (int j = 0; j < 4; ++j)
#pragma unroll
      for (int r = 0; r < 8; ++r) {
        const int m = mt * 32 + mi * 16 + mh * 8 + r;
        const int c = ng * 64 + j * 16 + nloc;
        const size_t idx = (size_t)m * Cdim + c;
        out1[idx] = x1a[idx] + acc[mi][j][r] + bias[c];
      }
}

// ---------------------------------------------------------------------------
extern "C" void kernel_launch(void* const* d_in, const int* in_sizes, int n_in,
                              void* d_out, int out_size, void* d_ws, size_t ws_size,
                              hipStream_t stream) {
  const float* x1     = (const float*)d_in[0];
  // d_in[1] = x2 (unused by the executed path)
  const float* n1g    = (const float*)d_in[2];
  const float* n1b    = (const float*)d_in[3];
  const float* qkv_w  = (const float*)d_in[4];
  const float* qkv_b  = (const float*)d_in[5];
  const float* proj_w = (const float*)d_in[6];
  const float* proj_b = (const float*)d_in[7];
  const float* rpb    = (const float*)d_in[8];
  const float* n2g    = (const float*)d_in[9];
  const float* n2b    = (const float*)d_in[10];
  const float* fc1_w  = (const float*)d_in[11];
  const float* fc1_b  = (const float*)d_in[12];
  const float* fc2_w  = (const float*)d_in[13];
  const float* fc2_b  = (const float*)d_in[14];

  // workspace carve-up (bf16 element offsets)
  bf16* w_qkv = (bf16*)d_ws;
  bf16* w_prj = w_qkv + (size_t)768 * 256;
  bf16* w_fc1 = w_prj + (size_t)256 * 256;
  bf16* w_fc2 = w_fc1 + (size_t)1024 * 256;
  bf16* xw    = w_fc2 + (size_t)256 * 1024;                       // 50176 x 256
  bf16* qb    = xw + (size_t)Mrows * Cdim;                        // 8192 x 64 x 32
  bf16* kb    = qb + (size_t)8192 * Npad * HDim;
  bf16* vTb   = kb + (size_t)8192 * Npad * HDim;
  bf16* yw    = vTb + (size_t)8192 * HDim * Npad;                 // 50176 x 256
  float* x1a  = (float*)(yw + (size_t)Mrows * Cdim);              // 50176 x 256 f32
  bf16* hbuf  = xw;   // reuse: xw dead after QKV GEMM
  bf16* hid   = qb;   // reuse: q/k/vT (+ start of yw) dead after proj

  float* out1 = (float*)d_out;                 // x1_new
  float* out2 = out1 + (size_t)Mrows * Cdim;   // out

  cvt_f32_bf16_kernel<<<768,  256, 0, stream>>>(qkv_w,  w_qkv, 768 * 256);
  cvt_f32_bf16_kernel<<<256,  256, 0, stream>>>(proj_w, w_prj, 256 * 256);
  cvt_f32_bf16_kernel<<<1024, 256, 0, stream>>>(fc1_w,  w_fc1, 1024 * 256);
  cvt_f32_bf16_kernel<<<1024, 256, 0, stream>>>(fc2_w,  w_fc2, 256 * 1024);
  // zero q/k/vT padded region (3 * 8192*64*32 bf16 = 100663296 B = 6291456 x 16B)
  zero16_kernel<<<24576, 256, 0, stream>>>((float4*)qb, 6291456L);

  ln1_shift_window_kernel<<<6272, 256, 0, stream>>>(x1, n1g, n1b, xw);
  qkv_gemm_kernel<<<2352, 256, 0, stream>>>(xw, w_qkv, qkv_b, qb, kb, vTb);
  attn_kernel<<<8192, 128, 0, stream>>>(qb, kb, vTb, rpb, yw);
  proj_gemm_kernel<<<784, 256, 0, stream>>>(yw, w_prj, proj_b, x1, x1a, out2);
  ln2_kernel<<<6272, 256, 0, stream>>>(x1a, n2g, n2b, hbuf);
  fc1_gemm_kernel<<<3136, 256, 0, stream>>>(hbuf, w_fc1, fc1_b, hid);
  fc2_gemm_kernel<<<784, 256, 0, stream>>>(hid, w_fc2, fc2_b, x1a, out1);
}
